// MultiHeadAttention_81484119540358
// MI455X (gfx1250) — matrix operand order repaired and checked by reading, not compiled, not run on hardware
//
#include <hip/hip_runtime.h>

#define E_DIM 2048
#define S_LEN 2048
#define B_SZ  2
#define H_NUM 32
#define D_HEAD 64
#define M_TOT (B_SZ * S_LEN)   // 4096

typedef __attribute__((ext_vector_type(16))) __bf16 bf16x16;
typedef __attribute__((ext_vector_type(8)))  float  f32x8;
typedef __attribute__((ext_vector_type(4)))  int    i32x4;

union FragU { uint4 u[2]; bf16x16 v; };

__device__ __forceinline__ unsigned short f32_to_bf16(float f) {
  unsigned int u = __float_as_uint(f);
  u += 0x7fffu + ((u >> 16) & 1u);   // round-to-nearest-even
  return (unsigned short)(u >> 16);
}

__device__ __forceinline__ f32x8 wmma_bf16(bf16x16 a, bf16x16 b, f32x8 c) {
  return __builtin_amdgcn_wmma_f32_16x16x32_bf16(false, a, false, b, (short)0, c,
                                                 false, false);
}

// ---------------------------------------------------------------------------
// CDNA5 async global->LDS (ASYNCcnt-tracked), with synchronous fallback.
// 16 bytes per lane per op. Builtin signature (from hipcc diagnostic):
//   arg0 = i32x4 addrspace(1)* (global, printed as "__device__"), then LDS
//   pointer, imm offset, imm cpol.
// ---------------------------------------------------------------------------
#if __has_builtin(__builtin_amdgcn_global_load_async_to_lds_b128)
#define HAVE_ASYNC_LDS 1
#else
#define HAVE_ASYNC_LDS 0
#endif

__device__ __forceinline__ void async_copy16(const unsigned short* gsrc,
                                             unsigned short* lds) {
#if HAVE_ASYNC_LDS
  __builtin_amdgcn_global_load_async_to_lds_b128(
      (__attribute__((address_space(1))) i32x4*)gsrc,
      (__attribute__((address_space(3))) i32x4*)lds, 0, 0);
#else
  *(uint4*)lds = *(const uint4*)gsrc;
#endif
}

__device__ __forceinline__ void wait_async_lds() {
#if HAVE_ASYNC_LDS
#if __has_builtin(__builtin_amdgcn_s_wait_asynccnt)
  __builtin_amdgcn_s_wait_asynccnt(0);
#else
  asm volatile("s_wait_asynccnt 0x0" ::: "memory");
#endif
#endif
}

// ---------------------------------------------------------------------------
// f32 -> bf16 convert, 8 elements per thread
// ---------------------------------------------------------------------------
__global__ void cvt_f32_to_bf16(const float* __restrict__ src,
                                unsigned short* __restrict__ dst, int n8) {
  int i = blockIdx.x * blockDim.x + threadIdx.x;
  if (i >= n8) return;
  const float4* s4 = (const float4*)src;
  float4 a = s4[2 * i], b = s4[2 * i + 1];
  uint4 o;
  o.x = (unsigned)f32_to_bf16(a.x) | ((unsigned)f32_to_bf16(a.y) << 16);
  o.y = (unsigned)f32_to_bf16(a.z) | ((unsigned)f32_to_bf16(a.w) << 16);
  o.z = (unsigned)f32_to_bf16(b.x) | ((unsigned)f32_to_bf16(b.y) << 16);
  o.w = (unsigned)f32_to_bf16(b.z) | ((unsigned)f32_to_bf16(b.w) << 16);
  ((uint4*)dst)[i] = o;
}

// ---------------------------------------------------------------------------
// GEMM: C[M_TOT, E] = A[M_TOT, E](bf16) @ W[E, E](bf16) + bias, epilogue:
//   optional head-wise RoPE (angle = h * 10000^(-j/32), pairs (d, d+32)),
//   MODE 0: bf16 out [B,H,S,D]   (Q, V)
//   MODE 1: bf16 out [B,H,D,S]   (K transposed for attention B-fragments)
//   MODE 2: f32  out [M_TOT, E]  (final projection)
// Block = 128 threads (4 waves, 2x2), block tile 128x128, wave tile 64x64.
// Double-buffered LDS tiles filled with async global->LDS loads; prefetch of
// tile k+1 overlaps the 16 WMMAs on tile k; s_wait_asynccnt 0 + barrier.
// ---------------------------------------------------------------------------
template <int MODE>
__global__ __launch_bounds__(128) void gemm128(
    const unsigned short* __restrict__ A, const unsigned short* __restrict__ W,
    const float* __restrict__ bias, void* __restrict__ outp, int applyRope) {
  __shared__ __align__(16) unsigned short As[2][128][40];  // 32 used, pad
  __shared__ __align__(16) unsigned short Bs[2][32][136];  // 128 used, pad

  const int t = threadIdx.x;
  const int lane = t & 31;
  const int wave = t >> 5;
  const int lh = lane & 15;
  const int hs = lane >> 4;
  const int m0 = blockIdx.y * 128;
  const int n0 = blockIdx.x * 128;
  const int mw = m0 + (wave >> 1) * 64;
  const int nw = n0 + (wave & 1) * 64;

  const int br = t >> 2, bc = (t & 3) * 32;
  const unsigned short* gA = A + (size_t)(m0 + t) * E_DIM;       // + k0
  const unsigned short* gB = W + (size_t)br * E_DIM + n0 + bc;   // + k0*E

  f32x8 acc[4][4];
#pragma unroll
  for (int i = 0; i < 4; ++i)
#pragma unroll
    for (int j = 0; j < 4; ++j)
#pragma unroll
      for (int e = 0; e < 8; ++e) acc[i][j][e] = 0.0f;

  // tile fill: A 128x32 (one 64B row per thread), B 32x128 (64B seg per thread)
  auto load_tile = [&](int k0, int buf) {
#pragma unroll
    for (int j = 0; j < 4; ++j)
      async_copy16(gA + k0 + j * 8, &As[buf][t][j * 8]);
#pragma unroll
    for (int j = 0; j < 4; ++j)
      async_copy16(gB + (size_t)k0 * E_DIM + j * 8, &Bs[buf][br][bc + j * 8]);
  };

  load_tile(0, 0);
  wait_async_lds();
  __syncthreads();

  for (int it = 0; it < E_DIM / 32; ++it) {
    const int cur = it & 1;
    if (it + 1 < E_DIM / 32) load_tile((it + 1) * 32, cur ^ 1);

    bf16x16 af[4], bf[4];
#pragma unroll
    for (int mr = 0; mr < 4; ++mr) {
      // 16-bit A 16x32 layout: lane l holds row (l&15), K = 8*(l>>4)+{0..7}, +16
      FragU f;
      const unsigned short* p = &As[cur][(wave >> 1) * 64 + mr * 16 + lh][hs * 8];
      f.u[0] = *(const uint4*)p;
      f.u[1] = *(const uint4*)(p + 16);
      af[mr] = f.v;
    }
#pragma unroll
    for (int nc = 0; nc < 4; ++nc) {
      // B 32x16 layout: lane l holds column (l&15), K = 8*(l>>4)+{0..7}, +16
      FragU g;
      const int col = (wave & 1) * 64 + nc * 16 + lh;
#pragma unroll
      for (int s = 0; s < 16; ++s) {
        const int kk = (s < 8) ? (hs * 8 + s) : (16 + hs * 8 + (s - 8));
        ((unsigned short*)g.u)[s] = Bs[cur][kk][col];
      }
      bf[nc] = g.v;
    }
#pragma unroll
    for (int mr = 0; mr < 4; ++mr)
#pragma unroll
      for (int nc = 0; nc < 4; ++nc)
        acc[mr][nc] = wmma_bf16(af[mr], bf[nc], acc[mr][nc]);

    wait_async_lds();   // prefetch of next tile fully in LDS
    __syncthreads();    // all waves done reading cur before it's overwritten
  }

  // ---- epilogue: bias, head-wise RoPE, store ----
  const int h = nw >> 6;  // wave covers exactly one head
#pragma unroll
  for (int nc = 0; nc < 4; ++nc) {
    float bv = bias[nw + nc * 16 + lh];
#pragma unroll
    for (int mr = 0; mr < 4; ++mr)
#pragma unroll
      for (int e = 0; e < 8; ++e) acc[mr][nc][e] += bv;
  }
  if (applyRope) {
#pragma unroll
    for (int c = 0; c < 2; ++c) {
      float j = (float)(c * 16 + lh);                          // j in [0,32)
      float invf = __expf(j * (-9.210340371976184f / 32.0f));  // 10000^(-j/32)
      float ang = (float)h * invf;
      float c0 = __cosf(ang), s0 = __sinf(ang);
#pragma unroll
      for (int mr = 0; mr < 4; ++mr)
#pragma unroll
        for (int e = 0; e < 8; ++e) {
          float xa = acc[mr][c][e], xb = acc[mr][c + 2][e];
          acc[mr][c][e]     = xa * c0 - xb * s0;
          acc[mr][c + 2][e] = xa * s0 + xb * c0;
        }
    }
  }
#pragma unroll
  for (int mr = 0; mr < 4; ++mr)
#pragma unroll
    for (int nc = 0; nc < 4; ++nc)
#pragma unroll
      for (int e = 0; e < 8; ++e) {
        int m = mw + mr * 16 + hs * 8 + e;   // C layout: row = 8*half + vgpr
        int n = nw + nc * 16 + lh;           // col = lane%16
        float v = acc[mr][nc][e];
        if (MODE == 2) {
          ((float*)outp)[(size_t)m * E_DIM + n] = v;
        } else {
          int bb = m >> 11, s = m & (S_LEN - 1);
          int hh = n >> 6, d = n & 63;
          size_t idx = (MODE == 0)
              ? (((size_t)(bb * H_NUM + hh)) * S_LEN + s) * D_HEAD + d
              : (((size_t)(bb * H_NUM + hh)) * D_HEAD + d) * S_LEN + s;
          ((unsigned short*)outp)[idx] = f32_to_bf16(v);
        }
      }
}

// ---------------------------------------------------------------------------
// Causal flash attention. One wave == one (b, h, 64-row q tile); 4 waves/block,
// fully independent (divergent kt trip counts, so no block barriers in loop).
// q:[B,H,S,D]  kT:[B,H,D,S]  v:[B,H,S,D]  -> out bf16 [B,S,E]
// ---------------------------------------------------------------------------
__global__ __launch_bounds__(128) void flash_attn(
    const unsigned short* __restrict__ qb, const unsigned short* __restrict__ kb,
    const unsigned short* __restrict__ vb, unsigned short* __restrict__ ob) {
  __shared__ __align__(16) unsigned short Ps[4][64][72];  // per-wave P tile

  const int lane = threadIdx.x & 31;
  const int w = threadIdx.x >> 5;
  const int lh = lane & 15, hs = lane >> 4;
  const int wid = blockIdx.x * 4 + w;
  const int bh = wid >> 5;   // 0..63 (b*H + h)
  const int qt = wid & 31;   // q tile index
  const int q0 = qt * 64;

  // Q fragments, resident for whole kernel (4 M-subtiles x 2 k-chunks of 32)
  bf16x16 qf[4][2];
#pragma unroll
  for (int mr = 0; mr < 4; ++mr)
#pragma unroll
    for (int ks = 0; ks < 2; ++ks) {
      FragU f;
      const unsigned short* p =
          qb + ((size_t)bh * S_LEN + q0 + mr * 16 + lh) * D_HEAD + ks * 32 + hs * 8;
      f.u[0] = *(const uint4*)p;
      f.u[1] = *(const uint4*)(p + 16);
      qf[mr][ks] = f.v;
    }

  f32x8 oacc[4][4];
#pragma unroll
  for (int i = 0; i < 4; ++i)
#pragma unroll
    for (int j = 0; j < 4; ++j)
#pragma unroll
      for (int e = 0; e < 8; ++e) oacc[i][j][e] = 0.0f;
  float mrow[4][8], lsum[4][8];
#pragma unroll
  for (int mr = 0; mr < 4; ++mr)
#pragma unroll
    for (int e = 0; e < 8; ++e) { mrow[mr][e] = -1e30f; lsum[mr][e] = 0.0f; }

  for (int kt = 0; kt <= qt; ++kt) {
    const int kbase = kt * 64;
    f32x8 sc[4][4];
#pragma unroll
    for (int i = 0; i < 4; ++i)
#pragma unroll
      for (int j = 0; j < 4; ++j)
#pragma unroll
        for (int e = 0; e < 8; ++e) sc[i][j][e] = 0.0f;

    // S = Q @ K^T  (K stored transposed: contiguous B-fragment rows)
#pragma unroll
    for (int ks = 0; ks < 2; ++ks) {
      bf16x16 kf[4];
#pragma unroll
      for (int nc = 0; nc < 4; ++nc) {
        FragU f;
#pragma unroll
        for (int s = 0; s < 16; ++s) {
          const int kk = (s < 8) ? (hs * 8 + s) : (16 + hs * 8 + (s - 8));
          ((unsigned short*)f.u)[s] =
              kb[((size_t)bh * D_HEAD + ks * 32 + kk) * S_LEN + kbase + nc * 16 + lh];
        }
        kf[nc] = f.v;
      }
#pragma unroll
      for (int mr = 0; mr < 4; ++mr)
#pragma unroll
        for (int nc = 0; nc < 4; ++nc)
          sc[mr][nc] = wmma_bf16(qf[mr][ks], kf[nc], sc[mr][nc]);
    }

    // scale 1/sqrt(64) + causal mask on the diagonal tile
    const bool diag = (kt == qt);
#pragma unroll
    for (int mr = 0; mr < 4; ++mr)
#pragma unroll
      for (int nc = 0; nc < 4; ++nc)
#pragma unroll
        for (int e = 0; e < 8; ++e) {
          float v = sc[mr][nc][e] * 0.125f;
          if (diag) {
            int qrow = q0 + mr * 16 + hs * 8 + e;
            int kcol = kbase + nc * 16 + lh;
            if (kcol > qrow) v = -1e30f;
          }
          sc[mr][nc][e] = v;
        }

    // online softmax; per-row stats live at fixed (half, vgpr); rows reduce
    // across the 16 lanes of each half (xor masks 1..8 never cross halves)
#pragma unroll
    for (int mr = 0; mr < 4; ++mr)
#pragma unroll
      for (int e = 0; e < 8; ++e) {
        float tm = fmaxf(fmaxf(sc[mr][0][e], sc[mr][1][e]),
                         fmaxf(sc[mr][2][e], sc[mr][3][e]));
        tm = fmaxf(tm, __shfl_xor(tm, 1));
        tm = fmaxf(tm, __shfl_xor(tm, 2));
        tm = fmaxf(tm, __shfl_xor(tm, 4));
        tm = fmaxf(tm, __shfl_xor(tm, 8));
        float nm = fmaxf(mrow[mr][e], tm);
        float corr = __expf(mrow[mr][e] - nm);
        mrow[mr][e] = nm;
        float ps = 0.0f;
#pragma unroll
        for (int nc = 0; nc < 4; ++nc) {
          float p = __expf(sc[mr][nc][e] - nm);
          sc[mr][nc][e] = p;
          ps += p;
          oacc[mr][nc][e] *= corr;
        }
        ps += __shfl_xor(ps, 1);
        ps += __shfl_xor(ps, 2);
        ps += __shfl_xor(ps, 4);
        ps += __shfl_xor(ps, 8);
        lsum[mr][e] = lsum[mr][e] * corr + ps;
      }

    // C-layout -> A-layout via per-wave LDS round trip (intra-wave; compiler
    // inserts s_wait_dscnt for the RAW)
#pragma unroll
    for (int mr = 0; mr < 4; ++mr)
#pragma unroll
      for (int nc = 0; nc < 4; ++nc)
#pragma unroll
        for (int e = 0; e < 8; ++e)
          Ps[w][mr * 16 + hs * 8 + e][nc * 16 + lh] = f32_to_bf16(sc[mr][nc][e]);

    // O += P @ V
#pragma unroll
    for (int ks = 0; ks < 2; ++ks) {
      bf16x16 vf[4];
#pragma unroll
      for (int nc = 0; nc < 4; ++nc) {
        FragU f;
#pragma unroll
        for (int s = 0; s < 16; ++s) {
          const int kk = (s < 8) ? (hs * 8 + s) : (16 + hs * 8 + (s - 8));
          ((unsigned short*)f.u)[s] =
              vb[((size_t)bh * S_LEN + kbase + ks * 32 + kk) * D_HEAD + nc * 16 + lh];
        }
        vf[nc] = f.v;
      }
      bf16x16 pf[4];
#pragma unroll
      for (int mr = 0; mr < 4; ++mr) {
        FragU f;
        const unsigned short* p = &Ps[w][mr * 16 + lh][ks * 32 + hs * 8];
        f.u[0] = *(const uint4*)p;
        f.u[1] = *(const uint4*)(p + 16);
        pf[mr] = f.v;
      }
#pragma unroll
      for (int mr = 0; mr < 4; ++mr)
#pragma unroll
        for (int nc = 0; nc < 4; ++nc)
          oacc[mr][nc] = wmma_bf16(pf[mr], vf[nc], oacc[mr][nc]);
    }
  }

  // finalize and store to [B,S,E] bf16
  const int bb = bh >> 5, hh = bh & 31;
#pragma unroll
  for (int mr = 0; mr < 4; ++mr)
#pragma unroll
    for (int e = 0; e < 8; ++e) {
      float inv = 1.0f / lsum[mr][e];
      int s = q0 + mr * 16 + hs * 8 + e;
      size_t base = ((size_t)bb * S_LEN + s) * E_DIM + hh * 64;
#pragma unroll
      for (int nc = 0; nc < 4; ++nc)
        ob[base + nc * 16 + lh] = f32_to_bf16(oacc[mr][nc][e] * inv);
    }
}

// ---------------------------------------------------------------------------
extern "C" void kernel_launch(void* const* d_in, const int* in_sizes, int n_in,
                              void* d_out, int out_size, void* d_ws, size_t ws_size,
                              hipStream_t stream) {
  (void)in_sizes; (void)n_in; (void)out_size; (void)ws_size;
  const float* x  = (const float*)d_in[0];
  const float* Wq = (const float*)d_in[1];
  const float* bq = (const float*)d_in[2];
  const float* Wk = (const float*)d_in[3];
  const float* bk = (const float*)d_in[4];
  const float* Wv = (const float*)d_in[5];
  const float* bv = (const float*)d_in[6];
  const float* Wo = (const float*)d_in[7];
  const float* bo = (const float*)d_in[8];
  float* out = (float*)d_out;

  char* ws = (char*)d_ws;
  size_t off = 0;
  auto carve = [&](size_t bytes) {
    char* p = ws + off;
    off += (bytes + 255) & ~(size_t)255;
    return p;
  };
  const size_t actB = (size_t)M_TOT * E_DIM * sizeof(unsigned short);  // 16.8 MB
  const size_t wB   = (size_t)E_DIM * E_DIM * sizeof(unsigned short);  //  8.4 MB
  unsigned short* xb   = (unsigned short*)carve(actB);
  unsigned short* wqb  = (unsigned short*)carve(wB);
  unsigned short* wkb  = (unsigned short*)carve(wB);
  unsigned short* wvb  = (unsigned short*)carve(wB);
  unsigned short* wob  = (unsigned short*)carve(wB);
  unsigned short* qb   = (unsigned short*)carve(actB);  // [B,H,S,D]
  unsigned short* kb   = (unsigned short*)carve(actB);  // [B,H,D,S]
  unsigned short* vb   = (unsigned short*)carve(actB);  // [B,H,S,D]
  unsigned short* obuf = (unsigned short*)carve(actB);  // [B,S,E]

  const int nx8 = M_TOT * E_DIM / 8;
  const int nw8 = E_DIM * E_DIM / 8;
  cvt_f32_to_bf16<<<(nx8 + 255) / 256, 256, 0, stream>>>(x, xb, nx8);
  cvt_f32_to_bf16<<<(nw8 + 255) / 256, 256, 0, stream>>>(Wq, wqb, nw8);
  cvt_f32_to_bf16<<<(nw8 + 255) / 256, 256, 0, stream>>>(Wk, wkb, nw8);
  cvt_f32_to_bf16<<<(nw8 + 255) / 256, 256, 0, stream>>>(Wv, wvb, nw8);
  cvt_f32_to_bf16<<<(nw8 + 255) / 256, 256, 0, stream>>>(Wo, wob, nw8);

  dim3 gg(E_DIM / 128, M_TOT / 128);  // (16, 32)
  gemm128<0><<<gg, 128, 0, stream>>>(xb, wqb, bq, qb, 1);  // Q + rope
  gemm128<1><<<gg, 128, 0, stream>>>(xb, wkb, bk, kb, 1);  // K + rope, transposed
  gemm128<0><<<gg, 128, 0, stream>>>(xb, wvb, bv, vb, 0);  // V

  flash_attn<<<(B_SZ * H_NUM * (S_LEN / 64)) / 4, 128, 0, stream>>>(qb, kb, vb, obuf);

  gemm128<2><<<gg, 128, 0, stream>>>(obuf, wob, bo, out, 0);  // out proj (f32)
}
